// LocalTokenMerger_47347719471649
// MI455X (gfx1250) — matrix operand order repaired and checked by reading, not compiled, not run on hardware
//
#include <hip/hip_runtime.h>
#include <hip/hip_bf16.h>

typedef __attribute__((ext_vector_type(16))) _Float16 v16h;
typedef __attribute__((ext_vector_type(8)))  float    v8f;

#define D_DIM 1024
#define G_DIM 64
#define T_LEN 8192
#define B_SZ  4
#define WIN   16

__device__ __forceinline__ void cvt8(v16h& dst, int o, float4 a, float4 b) {
  dst[o+0]=(_Float16)a.x; dst[o+1]=(_Float16)a.y; dst[o+2]=(_Float16)a.z; dst[o+3]=(_Float16)a.w;
  dst[o+4]=(_Float16)b.x; dst[o+5]=(_Float16)b.y; dst[o+6]=(_Float16)b.z; dst[o+7]=(_Float16)b.w;
}

// ---------------------------------------------------------------------------
// Kernel A: gn = normalize(relu(z @ w1^T) @ w2^T) per 16-token tile, 1 wave/tile.
// Stage 1: K=1024 in 32-chunks via v_wmma_f32_16x16x32_f16 (4 N-tiles of 16).
// Stage 2: C-layout -> LDS -> A-layout, K=64 in two 32-chunks.
// Stage 3: row-norm via cross-lane shfl_xor reduce inside 16-lane halves.
// ---------------------------------------------------------------------------
__global__ __launch_bounds__(128)
void proj_wmma_kernel(const float* __restrict__ z,
                      const float* __restrict__ w1,
                      const float* __restrict__ w2,
                      float* __restrict__ gn) {
  __shared__ float lds[4][16 * G_DIM];
  const int wave = threadIdx.x >> 5;
  const int lane = threadIdx.x & 31;
  const int tile = blockIdx.x * 4 + wave;
  const int row  = lane & 15;   // M index for A / N index for B
  const int half = lane >> 4;
  const long token0 = (long)tile * 16;
  const float* zrow = z + (token0 + row) * D_DIM;

  v8f acc[4] = {};
  for (int k0 = 0; k0 < D_DIM; k0 += 32) {
    // A fragment: 16x32 f16, lane row = token, halves cover K pairs.
    v16h a;
    {
      const float4 f0 = *(const float4*)(zrow + k0 + half * 8);
      const float4 f1 = *(const float4*)(zrow + k0 + half * 8 + 4);
      const float4 f2 = *(const float4*)(zrow + k0 + 16 + half * 8);
      const float4 f3 = *(const float4*)(zrow + k0 + 16 + half * 8 + 4);
      cvt8(a, 0, f0, f1);
      cvt8(a, 8, f2, f3);
    }
#pragma unroll
    for (int nt = 0; nt < 4; ++nt) {
      // B fragment: B[k,n] = w1[n, k]; lane n = row, half picks k 8h..8h+7 and 16+8h..16+8h+7.
      const float* wr = w1 + (long)(nt * 16 + row) * D_DIM + k0 + half * 8;
      const float4 g0 = *(const float4*)(wr + 0);
      const float4 g1 = *(const float4*)(wr + 4);
      const float4 g2 = *(const float4*)(wr + 16);
      const float4 g3 = *(const float4*)(wr + 20);
      v16h bf;
      cvt8(bf, 0, g0, g1);
      cvt8(bf, 8, g2, g3);
      acc[nt] = __builtin_amdgcn_wmma_f32_16x16x32_f16(
          false, a, false, bf, (short)0, acc[nt], false, false);
    }
  }

  // relu, spill g1 tile (16 x 64 f32) to LDS in (M, K) order for stage-2 A reads.
#pragma unroll
  for (int nt = 0; nt < 4; ++nt)
#pragma unroll
    for (int r = 0; r < 8; ++r) {
      float v = acc[nt][r];
      v = v > 0.f ? v : 0.f;
      lds[wave][(r + half * 8) * G_DIM + nt * 16 + row] = v;
    }
  __syncthreads();

  v8f acc2[4] = {};
  const float* lrow = &lds[wave][row * G_DIM];
#pragma unroll
  for (int kc = 0; kc < G_DIM; kc += 32) {
    v16h a2;
#pragma unroll
    for (int i = 0; i < 8; ++i) {
      a2[i]     = (_Float16)lrow[kc + half * 8 + i];
      a2[i + 8] = (_Float16)lrow[kc + 16 + half * 8 + i];
    }
#pragma unroll
    for (int nt = 0; nt < 4; ++nt) {
      // B2[k,h] = w2[h,k]
      const float* wr = w2 + (long)(nt * 16 + row) * G_DIM + kc + half * 8;
      const float4 g0 = *(const float4*)(wr + 0);
      const float4 g1 = *(const float4*)(wr + 4);
      const float4 g2 = *(const float4*)(wr + 16);
      const float4 g3 = *(const float4*)(wr + 20);
      v16h bf;
      cvt8(bf, 0, g0, g1);
      cvt8(bf, 8, g2, g3);
      acc2[nt] = __builtin_amdgcn_wmma_f32_16x16x32_f16(
          false, a2, false, bf, (short)0, acc2[nt], false, false);
    }
  }

  // Row norms: each accumulator slot r holds (M = r + 8*half, N = nt*16 + row).
#pragma unroll
  for (int r = 0; r < 8; ++r) {
    float s = 0.f;
#pragma unroll
    for (int nt = 0; nt < 4; ++nt) s += acc2[nt][r] * acc2[nt][r];
    s += __shfl_xor(s, 1, 32);
    s += __shfl_xor(s, 2, 32);
    s += __shfl_xor(s, 4, 32);
    s += __shfl_xor(s, 8, 32);
    const float inv = 1.0f / (sqrtf(s) + 1e-8f);
    const long tok = token0 + r + half * 8;
#pragma unroll
    for (int nt = 0; nt < 4; ++nt)
      gn[tok * G_DIM + nt * 16 + row] = acc2[nt][r] * inv;
  }
}

// ---------------------------------------------------------------------------
// Kernel B: sim[b,t] = <gn[b,t], gn[b,t+1]>, one thread per edge.
// ---------------------------------------------------------------------------
__global__ __launch_bounds__(256)
void sim_kernel(const float* __restrict__ gn, float* __restrict__ sim) {
  const int idx = blockIdx.x * blockDim.x + threadIdx.x;
  const int total = B_SZ * (T_LEN - 1);
  if (idx >= total) return;
  const int b = idx / (T_LEN - 1);
  const int t = idx % (T_LEN - 1);
  const float* p = gn + ((long)b * T_LEN + t) * G_DIM;
  const float* q = p + G_DIM;
  float s = 0.f;
#pragma unroll
  for (int i = 0; i < G_DIM; i += 4) {
    const float4 x = *(const float4*)(p + i);
    const float4 y = *(const float4*)(q + i);
    s += x.x * y.x + x.y * y.y + x.z * y.z + x.w * y.w;
  }
  sim[idx] = s;
}

// ---------------------------------------------------------------------------
// Kernel D: exclusive prefix of original token_lens per batch (for starts).
// ---------------------------------------------------------------------------
__global__ __launch_bounds__(256)
void prefix_kernel(const int* __restrict__ lens, float* __restrict__ pre) {
  __shared__ float part[256];
  const int b = blockIdx.x;
  const int* L = lens + (long)b * T_LEN;
  float* P = pre + (long)b * T_LEN;
  const int chunk = T_LEN / 256;
  const int s0 = threadIdx.x * chunk;
  float s = 0.f;
  for (int i = 0; i < chunk; ++i) s += (float)L[s0 + i];
  part[threadIdx.x] = s;
  __syncthreads();
  if (threadIdx.x == 0) {
    float run = 0.f;
    for (int i = 0; i < 256; ++i) { float v = part[i]; part[i] = run; run += v; }
  }
  __syncthreads();
  float run = part[threadIdx.x];
  for (int i = 0; i < chunk; ++i) { P[s0 + i] = run; run += (float)L[s0 + i]; }
}

// ---------------------------------------------------------------------------
// Kernel C: per-window top-k edge selection, merge & compact.
// out layout: [z_new (B*tl*D)] [lens_new (B*tl)] [starts_new (B*tl)], all f32.
// Compaction offset is analytic: out_base(w) = 16w - sum of clamped budgets.
// ---------------------------------------------------------------------------
__global__ __launch_bounds__(256)
void merge_kernel(const float* __restrict__ z, const int* __restrict__ lens,
                  const float* __restrict__ sim, const float* __restrict__ pre,
                  const int* __restrict__ p_tl, const int* __restrict__ p_off,
                  float* __restrict__ out) {
  const int w = blockIdx.x;
  const int b = blockIdx.y;
  const int tl = *p_tl;
  const int off = *p_off;
  const int nwin = T_LEN / WIN;

  __shared__ int sh_keptIdx[WIN];
  __shared__ int sh_mergedF[WIN];
  __shared__ int sh_nkept;
  __shared__ int sh_outbase;

  if (threadIdx.x == 0) {
    const int merges_needed = T_LEN - tl;
    int base = merges_needed / nwin;
    int extra = merges_needed % nwin;
    if (merges_needed <= 0) { base = 0; extra = 0; }
    const int p = off & 1;
    const int n_cand = (WIN - p) / 2;            // #allowed local edges in [0, WIN-2]
    const int cap = min(WIN / 2, n_cand);
    int budget = base + ((w < extra) ? 1 : 0);
    budget = min(budget, cap);
    if (budget < 0) budget = 0;

    float sc[WIN - 1];
    for (int l = 0; l < WIN - 1; ++l) {
      const int t = w * WIN + l;
      float v = (t < T_LEN - 1) ? sim[b * (T_LEN - 1) + t] : -__builtin_inff();
      if ((l & 1) != p) v = -__builtin_inff();
      sc[l] = v;
    }
    int mr = 0;  // bitmask: local token merged away
    for (int r = 0; r < budget; ++r) {
      int bi = -1; float bv = -__builtin_inff();
      for (int l = 0; l < WIN - 1; ++l)
        if (sc[l] > bv) { bv = sc[l]; bi = l; }   // strict > -> lowest-index ties
      if (bi < 0) break;                          // only -inf left
      sc[bi] = -__builtin_inff();
      mr |= 1 << (bi + 1);
    }
    int nk = 0;
    for (int l = 0; l < WIN; ++l) {
      if (!(mr & (1 << l))) {
        sh_keptIdx[nk] = l;
        sh_mergedF[nk] = (l + 1 < WIN) && (mr & (1 << (l + 1)));
        ++nk;
      }
    }
    sh_nkept = nk;
    const int b_hi = min(base + 1, cap);
    const int b_lo = min(base, cap);
    const int we = min(w, extra);
    sh_outbase = WIN * w - (b_hi * we + b_lo * (w - we));
  }
  __syncthreads();

  const int nk = sh_nkept;
  const int outbase = sh_outbase;
  float* out_z = out;
  float* out_lens = out + (long)B_SZ * tl * D_DIM;
  float* out_starts = out_lens + (long)B_SZ * tl;

  for (int e = 0; e < nk; ++e) {
    const int j = outbase + e;
    if (j < 0 || j >= tl) continue;
    const int l = sh_keptIdx[e];
    const int t = w * WIN + l;
    const int merged = sh_mergedF[e];
    const float wi = (float)lens[(long)b * T_LEN + t];
    const float wj = merged ? (float)lens[(long)b * T_LEN + t + 1] : 0.f;
    const float dsum = wi + wj;
    const float denom = dsum > 0.f ? dsum : 1.f;
    const float4* zi = (const float4*)(z + ((long)b * T_LEN + t) * D_DIM);
    const float4* zj = (const float4*)(z + ((long)b * T_LEN + t + 1) * D_DIM);
    float4* o = (float4*)(out_z + ((long)b * tl + j) * D_DIM);
    const int q = threadIdx.x;            // 256 threads * float4 == 1024 floats
    float4 x = zi[q];
    float4 r;
    if (merged) {
      const float4 y = zj[q];
      r.x = (x.x * wi + y.x * wj) / denom;
      r.y = (x.y * wi + y.y * wj) / denom;
      r.z = (x.z * wi + y.z * wj) / denom;
      r.w = (x.w * wi + y.w * wj) / denom;
    } else {
      r = x;
    }
    o[q] = r;
    if (threadIdx.x == 0) {
      out_lens[(long)b * tl + j] = dsum;
      out_starts[(long)b * tl + j] = pre[(long)b * T_LEN + t];
    }
  }
}

extern "C" void kernel_launch(void* const* d_in, const int* in_sizes, int n_in,
                              void* d_out, int out_size, void* d_ws, size_t ws_size,
                              hipStream_t stream) {
  const float* z    = (const float*)d_in[0];
  const int*   lens = (const int*)d_in[1];
  const float* w1   = (const float*)d_in[2];
  const float* w2   = (const float*)d_in[3];
  const int*   p_tl = (const int*)d_in[4];
  const int*   p_off= (const int*)d_in[5];
  float* out = (float*)d_out;

  char* ws = (char*)d_ws;
  float* gn  = (float*)ws;                                     // B*T*64 f32 = 8 MiB
  float* sim = (float*)(ws + (size_t)B_SZ * T_LEN * G_DIM * 4);// B*(T-1) f32
  float* pre = sim + (size_t)B_SZ * T_LEN;                     // B*T f32

  const int tiles = (B_SZ * T_LEN) / 16;                       // 2048 (one wave each)
  proj_wmma_kernel<<<tiles / 4, 128, 0, stream>>>(z, w1, w2, gn);

  const int simTotal = B_SZ * (T_LEN - 1);
  sim_kernel<<<(simTotal + 255) / 256, 256, 0, stream>>>(gn, sim);

  prefix_kernel<<<B_SZ, 256, 0, stream>>>(lens, pre);

  merge_kernel<<<dim3(T_LEN / WIN, B_SZ), 256, 0, stream>>>(
      z, lens, sim, pre, p_tl, p_off, out);
}
